// GRIN_59536836657977
// MI455X (gfx1250) — compile-verified, matrix operand order checked by reading, not run on hardware
//
#include <hip/hip_runtime.h>
#include <math.h>

#define NNODES 40000
#define NEDGES 640000
#define TSTEPS 8
#define HID    32
#define FEAT   16
#define NLAB   4
#define DIN    53   // FEAT + 1 + NLAB + HID
#define KPAD   64   // DIN padded to 2 WMMA k-chunks

typedef __attribute__((ext_vector_type(16))) _Float16 v16h;
typedef __attribute__((ext_vector_type(8)))  _Float16 v8h;
typedef __attribute__((ext_vector_type(8)))  float    v8f;
typedef int v4i_vec __attribute__((vector_size(16)));   // matches builtin param type

// ---- optional CDNA5 async global->LDS path (guarded; falls back cleanly) ---
#if defined(__has_builtin)
#if __has_builtin(__builtin_amdgcn_global_load_async_to_lds_b128) && \
    __has_builtin(__builtin_amdgcn_s_wait_asynccnt)
#define USE_ASYNC_LDS 1
#endif
#endif
#ifndef USE_ASYNC_LDS
#define USE_ASYNC_LDS 0
#endif

// ---------------- float <-> ordered-uint (for atomic max on floats) ---------
__device__ __forceinline__ unsigned fflip(float f) {
    unsigned u = __float_as_uint(f);
    return u ^ ((u & 0x80000000u) ? 0xFFFFFFFFu : 0x80000000u);
}
__device__ __forceinline__ float funflip(unsigned u) {
    return __uint_as_float(u ^ ((u & 0x80000000u) ? 0x80000000u : 0xFFFFFFFFu));
}

// ---------------- weight pre-pack into WMMA B-fragment layout ---------------
// 32 fragments: frag = ((c*2 + mat)*2 + chunk)*2 + nt, each 32 lanes x 16 halves.
// Element (lane, j): n = nt*16 + (lane&15); k = 32*chunk + j + 8*(lane>>4) + (j>=8 ? 8 : 0)
__global__ void pack_weights_kernel(const float* __restrict__ Wl,
                                    const float* __restrict__ Wr,
                                    _Float16* __restrict__ wfrag)
{
    int i = blockIdx.x * blockDim.x + threadIdx.x;   // 32 * 512 = 16384
    if (i >= 32 * 512) return;
    int frag = i >> 9, idx = i & 511;
    int lane = idx >> 4, j = idx & 15;
    int nt    = frag & 1;
    int chunk = (frag >> 1) & 1;
    int mat   = (frag >> 2) & 1;
    int c     = frag >> 3;
    int m = lane & 15, hh = lane >> 4;
    int n = nt * 16 + m;
    int k = 32 * chunk + j + 8 * hh + ((j >= 8) ? 8 : 0);
    const float* W = mat ? Wr : Wl;
    float v = (k < DIN) ? W[(c * HID + n) * DIN + k] : 0.0f;
    wfrag[i] = (_Float16)v;
}

// ---------------- WMMA GEMM: xl = inp @ Wl^T + bl ; xr = inp @ Wr^T + br ----
// inp_h: (N, 64) f16 (K-padded).  wfragC: this conv's 8 fragments (8*512 halves).
__global__ __launch_bounds__(128) void gemm_xlxr_kernel(
    const _Float16* __restrict__ inp_h,
    const _Float16* __restrict__ wfragC,
    const float* __restrict__ blv, const float* __restrict__ brv,
    float* __restrict__ xl, float* __restrict__ xr)
{
    __shared__ __attribute__((aligned(16))) _Float16 ldsA[4 * 16 * KPAD]; // 8 KB

    const int tid  = threadIdx.x;
    const int wave = tid >> 5;
    const int lane = tid & 31;
    const int tile = blockIdx.x * 4 + wave;  // N/16 = 2500 tiles, grid = 625

    // stage this wave's 16x64-half A tile (2 KB contiguous) into LDS
    {
        const uint4* srcp = (const uint4*)(inp_h + (size_t)tile * 16 * KPAD);
        uint4* dstp = (uint4*)(ldsA + (size_t)wave * 16 * KPAD);
#if USE_ASYNC_LDS
        #pragma unroll
        for (int i = 0; i < 4; ++i) {
            __builtin_amdgcn_global_load_async_to_lds_b128(
                (__attribute__((address_space(1))) v4i_vec*)(srcp + lane + 32 * i),
                (__attribute__((address_space(3))) v4i_vec*)(dstp + lane + 32 * i),
                0, 0);
        }
        __builtin_amdgcn_s_wait_asynccnt(0);
#else
        #pragma unroll
        for (int i = 0; i < 4; ++i) dstp[lane + 32 * i] = srcp[lane + 32 * i];
#endif
    }
    __syncthreads();

    const int m  = lane & 15;
    const int hh = lane >> 4;

    // A fragments: four contiguous 16B runs per lane (ds_load_b128)
    const _Float16* Aw = ldsA + (size_t)wave * 16 * KPAD + m * KPAD + 8 * hh;
    v8h a00 = *(const v8h*)(Aw + 0);    // chunk0, K = 8h .. 8h+7
    v8h a01 = *(const v8h*)(Aw + 16);   // chunk0, K = 16+8h ..
    v8h a10 = *(const v8h*)(Aw + 32);   // chunk1
    v8h a11 = *(const v8h*)(Aw + 48);
    v16h a0 = __builtin_shufflevector(a00, a01, 0,1,2,3,4,5,6,7,8,9,10,11,12,13,14,15);
    v16h a1 = __builtin_shufflevector(a10, a11, 0,1,2,3,4,5,6,7,8,9,10,11,12,13,14,15);

    // prepacked weight fragments: v16h per lane, coalesced global b128 loads
    const v16h* wf = (const v16h*)wfragC;        // frag f -> wf[f*32 + lane]
    v16h bl00 = wf[((0*2+0)*2+0) * 32 + lane];   // mat=L chunk0 nt0
    v16h bl01 = wf[((0*2+0)*2+1) * 32 + lane];   // mat=L chunk0 nt1
    v16h bl10 = wf[((0*2+1)*2+0) * 32 + lane];   // mat=L chunk1 nt0
    v16h bl11 = wf[((0*2+1)*2+1) * 32 + lane];
    v16h br00 = wf[((1*2+0)*2+0) * 32 + lane];
    v16h br01 = wf[((1*2+0)*2+1) * 32 + lane];
    v16h br10 = wf[((1*2+1)*2+0) * 32 + lane];
    v16h br11 = wf[((1*2+1)*2+1) * 32 + lane];

    v8f accL0 = {}, accL1 = {}, accR0 = {}, accR1 = {};
    accL0 = __builtin_amdgcn_wmma_f32_16x16x32_f16(false, a0, false, bl00, (short)0, accL0, false, false);
    accL0 = __builtin_amdgcn_wmma_f32_16x16x32_f16(false, a1, false, bl10, (short)0, accL0, false, false);
    accL1 = __builtin_amdgcn_wmma_f32_16x16x32_f16(false, a0, false, bl01, (short)0, accL1, false, false);
    accL1 = __builtin_amdgcn_wmma_f32_16x16x32_f16(false, a1, false, bl11, (short)0, accL1, false, false);
    accR0 = __builtin_amdgcn_wmma_f32_16x16x32_f16(false, a0, false, br00, (short)0, accR0, false, false);
    accR0 = __builtin_amdgcn_wmma_f32_16x16x32_f16(false, a1, false, br10, (short)0, accR0, false, false);
    accR1 = __builtin_amdgcn_wmma_f32_16x16x32_f16(false, a0, false, br01, (short)0, accR1, false, false);
    accR1 = __builtin_amdgcn_wmma_f32_16x16x32_f16(false, a1, false, br11, (short)0, accR1, false, false);

    // C/D layout: VGPR r -> row = r + 8*hh, col = m (plus n-tile offset)
    const int row0 = tile * 16;
    #pragma unroll
    for (int r = 0; r < 8; ++r) {
        int row = row0 + r + 8 * hh;
        xl[row * HID + m]      = accL0[r] + blv[m];
        xl[row * HID + 16 + m] = accL1[r] + blv[16 + m];
        xr[row * HID + m]      = accR0[r] + brv[m];
        xr[row * HID + 16 + m] = accR1[r] + brv[16 + m];
    }
}

// ---------------- edge pass 1: 8 threads/edge; seg-max ----------------------
__global__ void edge_max_kernel(const int* __restrict__ src, const int* __restrict__ dst,
                                const float* __restrict__ xl, const float* __restrict__ xr,
                                const float* __restrict__ att,
                                float* __restrict__ e, unsigned* __restrict__ mflip)
{
    int gid = blockIdx.x * blockDim.x + threadIdx.x;
    if (gid >= NEDGES * 8) return;
    int edge = gid >> 3, q = gid & 7;
    int s = src[edge], d = dst[edge];
    float4 l = ((const float4*)(xl + (size_t)s * HID))[q];
    float4 r = ((const float4*)(xr + (size_t)d * HID))[q];
    float4 a = ((const float4*)att)[q];
    float z, acc = 0.0f;
    z = l.x + r.x; z = (z > 0.0f) ? z : 0.2f * z; acc += z * a.x;
    z = l.y + r.y; z = (z > 0.0f) ? z : 0.2f * z; acc += z * a.y;
    z = l.z + r.z; z = (z > 0.0f) ? z : 0.2f * z; acc += z * a.z;
    z = l.w + r.w; z = (z > 0.0f) ? z : 0.2f * z; acc += z * a.w;
    acc += __shfl_xor(acc, 1, 32);
    acc += __shfl_xor(acc, 2, 32);
    acc += __shfl_xor(acc, 4, 32);
    if (q == 0) {
        e[edge] = acc;
        atomicMax(mflip + d, fflip(acc));
    }
}

// ---------------- edge pass 2: 8 threads/edge; den += a; raw += a*xl[src] ---
__global__ void edge_acc_kernel(const int* __restrict__ src, const int* __restrict__ dst,
                                const float* __restrict__ xl, const float* __restrict__ e,
                                const unsigned* __restrict__ mflip,
                                float* __restrict__ den, float* __restrict__ outraw)
{
    int gid = blockIdx.x * blockDim.x + threadIdx.x;
    if (gid >= NEDGES * 8) return;
    int edge = gid >> 3, q = gid & 7;
    int s = src[edge], d = dst[edge];
    float a = __expf(e[edge] - funflip(mflip[d]));
    if (q == 0) atomicAdd(den + d, a);
    float4 l = ((const float4*)(xl + (size_t)s * HID))[q];
    float* po = outraw + (size_t)d * HID + 4 * q;
    atomicAdd(po + 0, a * l.x);
    atomicAdd(po + 1, a * l.y);
    atomicAdd(po + 2, a * l.z);
    atomicAdd(po + 3, a * l.w);
}

// ---------------- node finalize: out = act(raw/den' + bconv) ----------------
// act: 0 = sigmoid, 1 = tanh, 2 = relu
__global__ void conv_finalize_kernel(const float* __restrict__ outraw,
                                     const float* __restrict__ den,
                                     const float* __restrict__ bconv,
                                     float* __restrict__ outbuf, int act)
{
    int i = blockIdx.x * blockDim.x + threadIdx.x;
    if (i >= NNODES * HID) return;
    int node = i >> 5, h = i & 31;
    float dn = den[node];
    dn = (dn > 0.0f) ? dn : 1.0f;
    float v = outraw[i] / dn + bconv[h];
    if (act == 0)      v = 1.0f / (1.0f + __expf(-v));
    else if (act == 1) v = tanhf(v);
    else               v = fmaxf(v, 0.0f);
    outbuf[i] = v;
}

// ---------------- assemble conv input directly as f16, K-padded to 64 -------
// mode 0/1: [p16(16) | mask(1) | labels(4) | pA(32) | 0-pad(11)]
// mode 2:   [p16(16) | mask(1) | pA(32)   | labels(4) | 0-pad(11)]
__global__ void assemble_kernel(const float* __restrict__ p16, const float* __restrict__ mask,
                                const float* __restrict__ labels, const float* __restrict__ pA,
                                int mode, _Float16* __restrict__ inp_h)
{
    int i = blockIdx.x * blockDim.x + threadIdx.x;
    if (i >= NNODES * KPAD) return;
    int node = i >> 6, col = i & 63;
    float v = 0.0f;
    if (col < DIN) {
        if (mode != 2) {
            if (col < FEAT)                 v = p16[node * FEAT + col];
            else if (col == FEAT)           v = mask[node];
            else if (col < FEAT + 1 + NLAB) v = labels[node * NLAB + (col - FEAT - 1)];
            else                            v = pA[node * HID + (col - FEAT - 1 - NLAB)];
        } else {
            if (col < FEAT)                 v = p16[node * FEAT + col];
            else if (col == FEAT)           v = mask[node];
            else if (col < FEAT + 1 + HID)  v = pA[node * HID + (col - FEAT - 1)];
            else                            v = labels[node * NLAB + (col - FEAT - 1 - HID)];
        }
    }
    inp_h[i] = (_Float16)v;
}

// ---------------- misc elementwise ------------------------------------------
__global__ void clear_kernel(unsigned* __restrict__ p, int n)
{
    int i = blockIdx.x * blockDim.x + threadIdx.x;
    if (i < n) p[i] = 0u;
}

__global__ void init_kernel(const float* __restrict__ x0, const float* __restrict__ mask,
                            float* __restrict__ hidden, float* __restrict__ x2)
{
    int i = blockIdx.x * blockDim.x + threadIdx.x;
    if (i < NNODES * HID) hidden[i] = 1.0f;
    if (i < NNODES * FEAT) {
        int node = i / FEAT;
        x2[i] = (mask[node] > 0.5f) ? x0[i] : 0.0f;
    }
}

__global__ void rh_kernel(const float* __restrict__ reset, const float* __restrict__ hidden,
                          float* __restrict__ rh)
{
    int i = blockIdx.x * blockDim.x + threadIdx.x;
    if (i < NNODES * HID) rh[i] = reset[i] * hidden[i];
}

__global__ void gate_kernel(float* __restrict__ hidden, const float* __restrict__ update,
                            const float* __restrict__ cell)
{
    int i = blockIdx.x * blockDim.x + threadIdx.x;
    if (i < NNODES * HID) {
        float u = update[i];
        hidden[i] = u * hidden[i] + (1.0f - u) * cell[i];
    }
}

// x1 = mask ? xt : hidden @ W1^T + b1
__global__ void y1_x1_kernel(const float* __restrict__ hidden, const float* __restrict__ W1,
                             const float* __restrict__ b1, const float* __restrict__ mask,
                             const float* __restrict__ xt, float* __restrict__ x1)
{
    int node = blockIdx.x * blockDim.x + threadIdx.x;
    if (node >= NNODES) return;
    float h[HID];
    const float4* ph = (const float4*)(hidden + (size_t)node * HID);
    #pragma unroll
    for (int q = 0; q < 8; ++q) {
        float4 v = ph[q];
        h[4 * q + 0] = v.x; h[4 * q + 1] = v.y; h[4 * q + 2] = v.z; h[4 * q + 3] = v.w;
    }
    bool mb = mask[node] > 0.5f;
    #pragma unroll
    for (int o = 0; o < FEAT; ++o) {
        float acc = b1[o];
        const float* w = W1 + o * HID;
        #pragma unroll
        for (int k = 0; k < HID; ++k) acc += h[k] * w[k];
        x1[node * FEAT + o] = mb ? xt[node * FEAT + o] : acc;
    }
}

// y2 = [s | hidden] @ W2^T + b2 ; x2 = mask ? xt : y2 ; out[t] = x2
__global__ void y2_x2_kernel(const float* __restrict__ sbuf, const float* __restrict__ hidden,
                             const float* __restrict__ W2, const float* __restrict__ b2,
                             const float* __restrict__ mask, const float* __restrict__ xt,
                             float* __restrict__ x2, float* __restrict__ outT)
{
    int node = blockIdx.x * blockDim.x + threadIdx.x;
    if (node >= NNODES) return;
    float sv[HID], hv[HID];
    const float4* ps = (const float4*)(sbuf + (size_t)node * HID);
    const float4* ph = (const float4*)(hidden + (size_t)node * HID);
    #pragma unroll
    for (int q = 0; q < 8; ++q) {
        float4 a = ps[q], b = ph[q];
        sv[4 * q + 0] = a.x; sv[4 * q + 1] = a.y; sv[4 * q + 2] = a.z; sv[4 * q + 3] = a.w;
        hv[4 * q + 0] = b.x; hv[4 * q + 1] = b.y; hv[4 * q + 2] = b.z; hv[4 * q + 3] = b.w;
    }
    bool mb = mask[node] > 0.5f;
    #pragma unroll
    for (int o = 0; o < FEAT; ++o) {
        float acc = b2[o];
        const float* w = W2 + o * (2 * HID);
        #pragma unroll
        for (int k = 0; k < HID; ++k) acc += sv[k] * w[k];
        #pragma unroll
        for (int k = 0; k < HID; ++k) acc += hv[k] * w[HID + k];
        float v = mb ? xt[node * FEAT + o] : acc;
        x2[node * FEAT + o] = v;
        outT[node * FEAT + o] = v;
    }
}

// ---------------- host-side conv driver -------------------------------------
static void run_conv(const _Float16* inp_h, const _Float16* wfragC,
                     const float* bl, const float* br,
                     const float* att, const float* bconv,
                     float* xl, float* xr, float* ebuf,
                     unsigned* mflip, float* den, float* outraw,
                     const int* src, const int* dst,
                     float* outbuf, int act, hipStream_t stream)
{
    gemm_xlxr_kernel<<<NNODES / 64, 128, 0, stream>>>(inp_h, wfragC, bl, br, xl, xr);
    clear_kernel<<<(NNODES * 34 + 255) / 256, 256, 0, stream>>>(mflip, NNODES * 34);
    edge_max_kernel<<<(NEDGES * 8 + 255) / 256, 256, 0, stream>>>(src, dst, xl, xr, att, ebuf, mflip);
    edge_acc_kernel<<<(NEDGES * 8 + 255) / 256, 256, 0, stream>>>(src, dst, xl, ebuf, mflip, den, outraw);
    conv_finalize_kernel<<<(NNODES * HID + 255) / 256, 256, 0, stream>>>(outraw, den, bconv, outbuf, act);
}

extern "C" void kernel_launch(void* const* d_in, const int* in_sizes, int n_in,
                              void* d_out, int out_size, void* d_ws, size_t ws_size,
                              hipStream_t stream)
{
    const float* x       = (const float*)d_in[0];   // (T, N, 16)
    const float* mask    = (const float*)d_in[1];   // (N,)
    const float* labels  = (const float*)d_in[2];   // (N, 4)
    const int*   eidx    = (const int*)  d_in[3];   // (2, E)
    // d_in[4] = edge_weight: unused by reference
    const float* Wl      = (const float*)d_in[5];   // (4, 32, 53)
    const float* bl      = (const float*)d_in[6];   // (4, 32)
    const float* Wr      = (const float*)d_in[7];
    const float* br      = (const float*)d_in[8];
    const float* att     = (const float*)d_in[9];   // (4, 32)
    const float* bconv   = (const float*)d_in[10];  // (4, 32)
    const float* W1      = (const float*)d_in[11];  // (16, 32)
    const float* b1      = (const float*)d_in[12];
    const float* W2      = (const float*)d_in[13];  // (16, 64)
    const float* b2      = (const float*)d_in[14];
    float* out = (float*)d_out;                     // (T, N, 16)

    const int* src = eidx;
    const int* dst = eidx + NEDGES;

    // ---- workspace carve-up (float units; ws base is >=256B aligned) ----
    float* ws = (float*)d_ws;
    size_t off = 0;
    _Float16* wfrag = (_Float16*)(ws + off); off += 32 * 512 / 2;        // 32 KB, aligned
    _Float16* inp_h = (_Float16*)(ws + off); off += (size_t)NNODES * KPAD / 2;
    float* hidden = ws + off; off += (size_t)NNODES * HID;
    float* x2     = ws + off; off += (size_t)NNODES * FEAT;
    float* xl     = ws + off; off += (size_t)NNODES * HID;
    float* xr     = ws + off; off += (size_t)NNODES * HID;
    float* rbuf   = ws + off; off += (size_t)NNODES * HID;  // reset
    float* ubuf   = ws + off; off += (size_t)NNODES * HID;  // update
    float* cbuf   = ws + off; off += (size_t)NNODES * HID;  // cell
    float* rh     = ws + off; off += (size_t)NNODES * HID;  // reset*hidden
    float* x1     = ws + off; off += (size_t)NNODES * FEAT;
    float* sbuf   = ws + off; off += (size_t)NNODES * HID;
    float* ebuf   = ws + off; off += (size_t)NEDGES;
    unsigned* mflip = (unsigned*)(ws + off); off += NNODES; // m/den/outraw contiguous
    float* den    = ws + off; off += NNODES;
    float* outraw = ws + off; off += (size_t)NNODES * HID;

    const int EW = 256;

    // one-time per launch: pack all conv weights into WMMA fragment layout
    pack_weights_kernel<<<(32 * 512 + EW - 1) / EW, EW, 0, stream>>>(Wl, Wr, wfrag);

    // hidden = 1; x2 = mask ? x[0] : 0
    init_kernel<<<(NNODES * HID + EW - 1) / EW, EW, 0, stream>>>(x, mask, hidden, x2);

    for (int t = 0; t < TSTEPS; ++t) {
        const float* xt = x + (size_t)t * NNODES * FEAT;

        // inp = [x2 | mask | labels | hidden]
        assemble_kernel<<<(NNODES * KPAD + EW - 1) / EW, EW, 0, stream>>>(x2, mask, labels, hidden, 0, inp_h);

        // reset = sigmoid(gatv2_0(inp)); update = sigmoid(gatv2_1(inp))
        run_conv(inp_h, wfrag + 0 * 8 * 512, bl + 0 * HID, br + 0 * HID,
                 att + 0 * HID, bconv + 0 * HID, xl, xr, ebuf, mflip, den, outraw,
                 src, dst, rbuf, /*sigmoid*/0, stream);
        run_conv(inp_h, wfrag + 1 * 8 * 512, bl + 1 * HID, br + 1 * HID,
                 att + 1 * HID, bconv + 1 * HID, xl, xr, ebuf, mflip, den, outraw,
                 src, dst, ubuf, /*sigmoid*/0, stream);

        // cin = [x2 | mask | labels | reset*hidden]; cell = tanh(gatv2_2(cin))
        rh_kernel<<<(NNODES * HID + EW - 1) / EW, EW, 0, stream>>>(rbuf, hidden, rh);
        assemble_kernel<<<(NNODES * KPAD + EW - 1) / EW, EW, 0, stream>>>(x2, mask, labels, rh, 1, inp_h);
        run_conv(inp_h, wfrag + 2 * 8 * 512, bl + 2 * HID, br + 2 * HID,
                 att + 2 * HID, bconv + 2 * HID, xl, xr, ebuf, mflip, den, outraw,
                 src, dst, cbuf, /*tanh*/1, stream);

        // hidden = u*hidden + (1-u)*cell
        gate_kernel<<<(NNODES * HID + EW - 1) / EW, EW, 0, stream>>>(hidden, ubuf, cbuf);

        // x1 = mask ? xt : hidden @ W1^T + b1
        y1_x1_kernel<<<(NNODES + EW - 1) / EW, EW, 0, stream>>>(hidden, W1, b1, mask, xt, x1);

        // ft2 = [x1 | mask | hidden | labels]; s = relu(gatv2_3(ft2))
        assemble_kernel<<<(NNODES * KPAD + EW - 1) / EW, EW, 0, stream>>>(x1, mask, labels, hidden, 2, inp_h);
        run_conv(inp_h, wfrag + 3 * 8 * 512, bl + 3 * HID, br + 3 * HID,
                 att + 3 * HID, bconv + 3 * HID, xl, xr, ebuf, mflip, den, outraw,
                 src, dst, sbuf, /*relu*/2, stream);

        // y2 = [s|hidden] @ W2^T + b2; x2 = mask ? xt : y2; out[t] = x2
        y2_x2_kernel<<<(NNODES + EW - 1) / EW, EW, 0, stream>>>(sbuf, hidden, W2, b2, mask, xt,
                                                               x2, out + (size_t)t * NNODES * FEAT);
    }
}
